// GraphConvolution_59107339927854
// MI455X (gfx1250) — matrix operand order repaired and checked by reading, not compiled, not run on hardware
//
#include <hip/hip_runtime.h>

// GCN layer: out = leaky_relu( D^-1/2 A D^-1/2 (XW) + XW + b )
// N=16384, B=2, D_IN=D_OUT=32, fp32 in/out.
//
// MI455X plan:
//  - adj (1 GiB fp32) read twice (rowsum pass + GEMM pass): ~92us HBM floor @23.3TB/s.
//  - big GEMM agg = norm_adj @ Yp done with v_wmma_f32_16x16x32_bf16 using a
//    bf16x3 error-compensated split (a_hi*b_hi + a_lo*b_hi + a_hi*b_lo, fp32 acc)
//    -> near-fp32 accuracy while staying on the fast bf16 WMMA path; still
//    memory-bound on the adj stream.
//  - s[j] folded into Yp[j] = s[j]*(X@W)[j]; s[i] applied in the epilogue.

#define N_NODES   16384
#define BATCH     2
#define D_IN      32
#define D_OUT     32
#define NEG_SLOPE 0.01f

typedef __attribute__((ext_vector_type(16))) __bf16 v16bf;
typedef __attribute__((ext_vector_type(8)))  float  v8f;

// ---------------------------------------------------------------------------
// Kernel 1: row sums of adj -> inv_sqrt[n] = rsqrt(sum_j adj[n,j])
// One wave32 per row, float4 streaming, shuffle reduction.
// ---------------------------------------------------------------------------
__global__ __launch_bounds__(256) void rowsum_kernel(const float* __restrict__ adj,
                                                     float* __restrict__ inv_sqrt) {
    const int wave = threadIdx.x >> 5;
    const int lane = threadIdx.x & 31;
    const int row  = blockIdx.x * 8 + wave;
    const float4* p = (const float4*)(adj + (size_t)row * N_NODES);
    float acc = 0.f;
    for (int i = lane; i < N_NODES / 4; i += 32) {
        float4 v = p[i];
        acc += (v.x + v.y) + (v.z + v.w);
    }
#pragma unroll
    for (int off = 16; off > 0; off >>= 1) acc += __shfl_xor(acc, off, 32);
    if (lane == 0) inv_sqrt[row] = rsqrtf(acc);
}

// ---------------------------------------------------------------------------
// Kernel 2: Xp = X @ W (fp32, kept for residual), and
//           YpT_hi/lo[b][o][n] = bf16 hi/lo split of s[n] * Xp[b][n][o]
// YpT is stored transposed so the WMMA B-fragment is a contiguous 32B load.
// ---------------------------------------------------------------------------
__global__ __launch_bounds__(256) void project_kernel(const float* __restrict__ X,
                                                      const float* __restrict__ W,
                                                      const float* __restrict__ inv_sqrt,
                                                      float* __restrict__ Xp,
                                                      __bf16* __restrict__ ypt_hi,
                                                      __bf16* __restrict__ ypt_lo) {
    __shared__ float Wl[D_IN * D_OUT];
    const int tid = threadIdx.x;
    for (int i = tid; i < D_IN * D_OUT; i += 256) Wl[i] = W[i];
    __syncthreads();

    const int gid = blockIdx.x * 256 + tid;    // 0 .. B*N-1
    const int b   = gid >> 14;                 // gid / N_NODES
    const int n   = gid & (N_NODES - 1);

    float x[D_IN];
    const float4* x4 = (const float4*)(X + (size_t)gid * D_IN);
#pragma unroll
    for (int i = 0; i < D_IN / 4; ++i) {
        float4 v = x4[i];
        x[4 * i + 0] = v.x; x[4 * i + 1] = v.y;
        x[4 * i + 2] = v.z; x[4 * i + 3] = v.w;
    }
    const float s = inv_sqrt[n];

#pragma unroll 4
    for (int o = 0; o < D_OUT; ++o) {
        float a = 0.f;
#pragma unroll
        for (int i = 0; i < D_IN; ++i) a = fmaf(x[i], Wl[i * D_OUT + o], a);
        Xp[(size_t)gid * D_OUT + o] = a;
        const float  y = s * a;
        const __bf16 h = (__bf16)y;
        const size_t off = (size_t)(b * D_OUT + o) * N_NODES + n;
        ypt_hi[off] = h;
        ypt_lo[off] = (__bf16)(y - (float)h);
    }
}

// ---------------------------------------------------------------------------
// Kernel 3: agg = adj @ Yp via bf16 WMMA (bf16x3), epilogue fuses
//           s[i] scale + Xp residual + bias + LeakyReLU.
// ---------------------------------------------------------------------------

// Convert a float4 into elements [E..E+3] of the hi/lo bf16 A fragments.
template <int E>
static __device__ __forceinline__ void split4(float4 v, v16bf& hi, v16bf& lo) {
    const float f0 = v.x, f1 = v.y, f2 = v.z, f3 = v.w;
    const __bf16 h0 = (__bf16)f0, h1 = (__bf16)f1, h2 = (__bf16)f2, h3 = (__bf16)f3;
    hi[E + 0] = h0; hi[E + 1] = h1; hi[E + 2] = h2; hi[E + 3] = h3;
    lo[E + 0] = (__bf16)(f0 - (float)h0);
    lo[E + 1] = (__bf16)(f1 - (float)h1);
    lo[E + 2] = (__bf16)(f2 - (float)h2);
    lo[E + 3] = (__bf16)(f3 - (float)h3);
}

#define WMMA_BF16(A, B, C) \
    __builtin_amdgcn_wmma_f32_16x16x32_bf16(false, (A), false, (B), (short)0, (C), false, false)

__global__ __launch_bounds__(256) void gcn_wmma_kernel(const float*  __restrict__ adj,
                                                       const float*  __restrict__ inv_sqrt,
                                                       const float*  __restrict__ Xp,
                                                       const __bf16* __restrict__ ypt_hi,
                                                       const __bf16* __restrict__ ypt_lo,
                                                       const float*  __restrict__ bias,
                                                       float* __restrict__ out) {
    const int tid  = threadIdx.x;
    const int wave = tid >> 5;
    const int lane = tid & 31;
    const int half = lane >> 4;   // lane group 0-15 / 16-31
    const int m    = lane & 15;
    const int row0 = blockIdx.x * 128 + wave * 16;   // 16-row stripe per wave

    // 4 output tiles of 16x16: (batch 0/1) x (cols 0-15 / 16-31), fp32 accum.
    v8f acc[4] = {};

    // A-fragment source: lane (half,m) holds row m of the stripe, K-run
    // [half*16 .. half*16+15] (contiguous, matching the B fragment).
    const float* aptr = adj + (size_t)(row0 + m) * N_NODES + half * 16;

    // B-fragment source: lane = column n (=m), K-run of 16 per lane half,
    // consecutive-K bf16 pairs packed per dword -> contiguous in YpT[b][o][.].
    size_t boff[4];
#pragma unroll
    for (int t = 0; t < 4; ++t) {
        const int bt = t >> 1, nt = t & 1;
        boff[t] = (size_t)(bt * D_OUT + nt * 16 + m) * N_NODES + (size_t)(half * 16);
    }

    for (int k0 = 0; k0 < N_NODES; k0 += 32) {
        // 16 fp32 of this lane's adj row (one 16-element run) = 4x b128.
        float4 r0a = *(const float4*)(aptr + k0);
        float4 r0b = *(const float4*)(aptr + k0 + 4);
        float4 r1a = *(const float4*)(aptr + k0 + 8);
        float4 r1b = *(const float4*)(aptr + k0 + 12);

        if (k0 + 512 < N_NODES)                     // stream-ahead on the adj row
            __builtin_prefetch(aptr + k0 + 512, 0, 0);

        v16bf ahi, alo;
        split4<0>(r0a, ahi, alo);
        split4<4>(r0b, ahi, alo);
        split4<8>(r1a, ahi, alo);
        split4<12>(r1b, ahi, alo);

#pragma unroll
        for (int t = 0; t < 4; ++t) {
            const v16bf bhi = *(const v16bf*)(ypt_hi + boff[t] + k0);
            const v16bf blo = *(const v16bf*)(ypt_lo + boff[t] + k0);
            acc[t] = WMMA_BF16(ahi, bhi, acc[t]);   // main product
            acc[t] = WMMA_BF16(alo, bhi, acc[t]);   // A residual correction
            acc[t] = WMMA_BF16(ahi, blo, acc[t]);   // B residual correction
        }
    }

    // Epilogue: out = leaky_relu( s[i]*acc + Xp + bias )
    // C/D layout: VGPR r -> row (half*8 + r), lane m -> column.
    float sr[8];
#pragma unroll
    for (int r = 0; r < 8; ++r) sr[r] = inv_sqrt[row0 + half * 8 + r];

#pragma unroll
    for (int t = 0; t < 4; ++t) {
        const int bt = t >> 1, nt = t & 1;
        const int col = nt * 16 + m;
        const float bv = bias[col];
#pragma unroll
        for (int r = 0; r < 8; ++r) {
            const int row = row0 + half * 8 + r;
            const size_t o = ((size_t)bt * N_NODES + row) * D_OUT + col;
            float v = sr[r] * acc[t][r] + Xp[o] + bv;
            out[o] = (v >= 0.f) ? v : NEG_SLOPE * v;
        }
    }
}

// ---------------------------------------------------------------------------
// Launch: rowsum -> project -> wmma GEMM (all on `stream`, serialized).
// Workspace layout (needs ~8.1 MiB):
//   [0)            inv_sqrt : N fp32            (64 KiB)
//   [64K)          Xp       : B*N*32 fp32       (4 MiB)
//   [64K+4M)       YpT_hi   : B*32*N bf16       (2 MiB)
//   [64K+6M)       YpT_lo   : B*32*N bf16       (2 MiB)
// ---------------------------------------------------------------------------
extern "C" void kernel_launch(void* const* d_in, const int* in_sizes, int n_in,
                              void* d_out, int out_size, void* d_ws, size_t ws_size,
                              hipStream_t stream) {
    (void)in_sizes; (void)n_in; (void)out_size; (void)ws_size;
    const float* adj = (const float*)d_in[0];
    const float* X   = (const float*)d_in[1];
    const float* W   = (const float*)d_in[2];
    const float* b   = (const float*)d_in[3];
    float* out = (float*)d_out;

    char* ws = (char*)d_ws;
    float*  inv_sqrt = (float*)(ws);
    float*  Xp       = (float*)(ws + 65536);
    __bf16* ypt_hi   = (__bf16*)(ws + 65536 + 4194304);
    __bf16* ypt_lo   = (__bf16*)(ws + 65536 + 4194304 + 2097152);

    rowsum_kernel<<<N_NODES / 8, 256, 0, stream>>>(adj, inv_sqrt);
    project_kernel<<<(BATCH * N_NODES) / 256, 256, 0, stream>>>(X, W, inv_sqrt,
                                                                Xp, ypt_hi, ypt_lo);
    gcn_wmma_kernel<<<N_NODES / 128, 256, 0, stream>>>(adj, inv_sqrt, Xp,
                                                       ypt_hi, ypt_lo, b, out);
}
